// GRUDecoder_28011776704978
// MI455X (gfx1250) — matrix operand order repaired and checked by reading, not compiled, not run on hardware
//
#include <hip/hip_runtime.h>
#include <hip/hip_bf16.h>
#include <math.h>

#define B_  512
#define T_  32
#define H_  2048
#define E_  4096
#define G3_ 6144   // 3*H

typedef __attribute__((ext_vector_type(16))) __bf16       v16bf;
typedef __attribute__((ext_vector_type(8)))  float        v8f;
typedef __attribute__((ext_vector_type(4)))  unsigned int v4u;

__device__ __forceinline__ unsigned short f2bf(float f) {
    unsigned int u = __builtin_bit_cast(unsigned int, f);
    u += 0x7FFFu + ((u >> 16) & 1u);          // round-to-nearest-even
    return (unsigned short)(u >> 16);
}

union Frag { v16bf v; v4u q[2]; };

#define BM   128
#define BN   128
#define BK   32
#define NT   128   // threads per block = 4 waves (2x2 wave grid, 64x64 per wave)
#define LSTR 40    // ushorts per LDS row (80 B: 16B-aligned, bank-spread)

// C[m,n] = sum_k A[m,k]*W[n,k] + bias[n]; A (M x K) bf16 row-major stride lda,
// W (N x K) bf16 row-major stride K, C fp32 stride ldc. grid = (N/BN, M/BM).
__global__ __launch_bounds__(NT) void gemm_bf16_wmma(
    const unsigned short* __restrict__ A, int lda,
    const unsigned short* __restrict__ W,
    const float* __restrict__ bias,
    float* __restrict__ C, int ldc, int K)
{
    __shared__ __align__(16) unsigned short sA[BM * LSTR];
    __shared__ __align__(16) unsigned short sB[BN * LSTR];

    const int tid  = threadIdx.x;
    const int lane = tid & 31;
    const int wv   = tid >> 5;         // 0..3
    const int wr   = wv >> 1;          // 0..1 : 64-row slab
    const int wc   = wv & 1;           // 0..1 : 64-col slab
    const int l16  = lane & 15;
    const int half = lane >> 4;        // 0/1
    const int mb   = blockIdx.y * BM;
    const int nb   = blockIdx.x * BN;

    v8f acc[4][4] = {};

    for (int kk = 0; kk < K; kk += BK) {
        __syncthreads();
        // stage 128x32 A tile + 128x32 W tile: 2*512 b128 slots, 8 per thread
        #pragma unroll
        for (int s = 0; s < 4; ++s) {
            int slot = tid + s * NT;
            int row  = slot >> 2, c8 = slot & 3;
            v4u av = *(const v4u*)(A + (size_t)(mb + row) * lda + kk + c8 * 8);
            *(v4u*)(sA + row * LSTR + c8 * 8) = av;
            v4u wvv = *(const v4u*)(W + (size_t)(nb + row) * (size_t)K + kk + c8 * 8);
            *(v4u*)(sB + row * LSTR + c8 * 8) = wvv;
        }
        __syncthreads();

        // A frag: lane holds row M=l16, K = 8*half + {0..7, 16..23}
        Frag a[4], b[4];
        #pragma unroll
        for (int i = 0; i < 4; ++i) {
            const unsigned short* p = sA + (wr * 64 + i * 16 + l16) * LSTR;
            a[i].q[0] = *(const v4u*)(p + half * 8);
            a[i].q[1] = *(const v4u*)(p + 16 + half * 8);
        }
        // B frag: lane holds col N=l16, K = 8*half + {0..7, 16..23}
        #pragma unroll
        for (int j = 0; j < 4; ++j) {
            const unsigned short* p = sB + (wc * 64 + j * 16 + l16) * LSTR;
            b[j].q[0] = *(const v4u*)(p + half * 8);
            b[j].q[1] = *(const v4u*)(p + 16 + half * 8);
        }
        #pragma unroll
        for (int i = 0; i < 4; ++i)
            #pragma unroll
            for (int j = 0; j < 4; ++j)
                acc[i][j] = __builtin_amdgcn_wmma_f32_16x16x32_bf16(
                    false, a[i].v, false, b[j].v, (short)0, acc[i][j], false, false);
    }

    // C/D layout: VGPR r, lanes 0-15 -> M=r, lanes 16-31 -> M=8+r; N = l16
    #pragma unroll
    for (int i = 0; i < 4; ++i)
        #pragma unroll
        for (int j = 0; j < 4; ++j) {
            const int   nc = nb + wc * 64 + j * 16 + l16;
            const float bv = bias[nc];
            #pragma unroll
            for (int r = 0; r < 8; ++r) {
                const int mr = mb + wr * 64 + i * 16 + half * 8 + r;
                C[(size_t)mr * ldc + nc] = acc[i][j][r] + bv;
            }
        }
}

__global__ __launch_bounds__(256) void cvt_bf16(const float* __restrict__ s,
                                                unsigned short* __restrict__ d, int n) {
    for (int i = blockIdx.x * 256 + threadIdx.x; i < n; i += gridDim.x * 256)
        d[i] = f2bf(s[i]);
}

__global__ __launch_bounds__(256) void init_h(const float* __restrict__ ctx,
                                              float* __restrict__ h,
                                              unsigned short* __restrict__ hbf) {
    int i = blockIdx.x * 256 + threadIdx.x;   // B*H threads
    float v = ctx[i];
    h[i] = v;
    hbf[i] = f2bf(v);
}

// per-step teacher input -> bf16 (fallback path)
__global__ __launch_bounds__(256) void x_prep(const float* __restrict__ tgt,
                                              const float* __restrict__ sos,
                                              unsigned short* __restrict__ xb, int t) {
    int idx = blockIdx.x * 256 + threadIdx.x; // B*E threads
    int b = idx >> 12, e = idx & (E_ - 1);
    float v = (t == 0) ? sos[e] : tgt[((size_t)b * T_ + (t - 1)) * E_ + e];
    xb[idx] = f2bf(v);
}

// all-steps teacher input -> bf16, row m = b*T + t  (hoisted path)
__global__ __launch_bounds__(256) void x_all_prep(const float* __restrict__ tgt,
                                                  const float* __restrict__ sos,
                                                  unsigned short* __restrict__ xall) {
    const size_t n = (size_t)B_ * T_ * E_;
    for (size_t idx = (size_t)blockIdx.x * 256 + threadIdx.x; idx < n;
         idx += (size_t)gridDim.x * 256) {
        int e  = (int)(idx & (E_ - 1));
        size_t bt = idx >> 12;
        int t  = (int)(bt & (T_ - 1));
        int b  = (int)(bt >> 5);
        float v = (t == 0) ? sos[e] : tgt[((size_t)b * T_ + (t - 1)) * E_ + e];
        xall[idx] = f2bf(v);
    }
}

__global__ __launch_bounds__(256) void gru_step(const float* __restrict__ gx,
                                                int gx_stride,
                                                const float* __restrict__ gh,
                                                float* __restrict__ h,
                                                unsigned short* __restrict__ hbf) {
    int idx = blockIdx.x * 256 + threadIdx.x; // B*H threads
    int b = idx >> 11, j = idx & (H_ - 1);
    const float* px = gx + (size_t)b * gx_stride;
    const float* ph = gh + (size_t)b * G3_;
    float r = 1.0f / (1.0f + expf(-(px[j]      + ph[j])));
    float z = 1.0f / (1.0f + expf(-(px[H_ + j] + ph[H_ + j])));
    float n = tanhf(px[2 * H_ + j] + r * ph[2 * H_ + j]);
    float hv = h[idx];
    float hn = (1.0f - z) * n + z * hv;
    h[idx]  = hn;
    hbf[idx] = f2bf(hn);
}

// in-place log-softmax over one row of E=4096, 256 threads, 16 vals/thread
__global__ __launch_bounds__(256) void log_softmax_row(float* __restrict__ out, int t) {
    const int b   = blockIdx.x;
    const int tid = threadIdx.x;
    float* row = out + ((size_t)b * T_ + t) * E_;
    float v[16];
    float mx = -INFINITY;
    #pragma unroll
    for (int i = 0; i < 16; ++i) { v[i] = row[tid + i * 256]; mx = fmaxf(mx, v[i]); }

    __shared__ float red[256];
    red[tid] = mx; __syncthreads();
    for (int s = 128; s > 0; s >>= 1) {
        if (tid < s) red[tid] = fmaxf(red[tid], red[tid + s]);
        __syncthreads();
    }
    mx = red[0]; __syncthreads();

    float sm = 0.0f;
    #pragma unroll
    for (int i = 0; i < 16; ++i) sm += expf(v[i] - mx);
    red[tid] = sm; __syncthreads();
    for (int s = 128; s > 0; s >>= 1) {
        if (tid < s) red[tid] += red[tid + s];
        __syncthreads();
    }
    const float lse = mx + logf(red[0]);
    #pragma unroll
    for (int i = 0; i < 16; ++i) row[tid + i * 256] = v[i] - lse;
}

extern "C" void kernel_launch(void* const* d_in, const int* in_sizes, int n_in,
                              void* d_out, int out_size, void* d_ws, size_t ws_size,
                              hipStream_t stream) {
    const float* ctx   = (const float*)d_in[0];
    const float* tgt   = (const float*)d_in[1];
    const float* sos   = (const float*)d_in[2];
    const float* W_ih  = (const float*)d_in[3];
    const float* W_hh  = (const float*)d_in[4];
    const float* b_ih  = (const float*)d_in[5];
    const float* b_hh  = (const float*)d_in[6];
    const float* W_out = (const float*)d_in[7];
    const float* b_out = (const float*)d_in[8];
    float* out = (float*)d_out;

    char* p = (char*)d_ws;
    size_t used = 0;
    auto take = [&](size_t bytes) {
        char* r = p;
        size_t a = (bytes + 255) & ~(size_t)255;
        p += a; used += a;
        return (void*)r;
    };

    // common allocations
    unsigned short* Wih_bf  = (unsigned short*)take((size_t)G3_ * E_ * 2); // 50.3 MB
    unsigned short* Whh_bf  = (unsigned short*)take((size_t)G3_ * H_ * 2); // 25.2 MB
    unsigned short* Wout_bf = (unsigned short*)take((size_t)E_  * H_ * 2); // 16.8 MB
    unsigned short* h_bf    = (unsigned short*)take((size_t)B_ * H_ * 2);  //  2.1 MB
    float*          h       = (float*)take((size_t)B_ * H_ * 4);           //  4.2 MB
    float*          gh      = (float*)take((size_t)B_ * G3_ * 4);          // 12.6 MB

    const size_t big_bytes =
        ((((size_t)B_ * T_ * E_ * 2) + 255) & ~(size_t)255) +    // x_all bf16 134 MB
        ((((size_t)B_ * T_ * G3_ * 4) + 255) & ~(size_t)255);    // gx_all f32 403 MB
    const bool hoist = (ws_size >= used + big_bytes);

    unsigned short* x_bf  = nullptr;  // per-step
    float*          gx    = nullptr;  // per-step
    unsigned short* x_all = nullptr;  // hoisted
    float*          gx_all= nullptr;  // hoisted
    if (hoist) {
        x_all  = (unsigned short*)take((size_t)B_ * T_ * E_ * 2);
        gx_all = (float*)take((size_t)B_ * T_ * G3_ * 4);
    } else {
        x_bf = (unsigned short*)take((size_t)B_ * E_ * 2);
        gx   = (float*)take((size_t)B_ * G3_ * 4);
    }

    // one-time: bf16 weights + initial hidden state
    cvt_bf16<<<4096, 256, 0, stream>>>(W_ih,  Wih_bf,  G3_ * E_);
    cvt_bf16<<<4096, 256, 0, stream>>>(W_hh,  Whh_bf,  G3_ * H_);
    cvt_bf16<<<4096, 256, 0, stream>>>(W_out, Wout_bf, E_  * H_);
    init_h<<<(B_ * H_) / 256, 256, 0, stream>>>(ctx, h, h_bf);

    if (hoist) {
        // gx for ALL timesteps in one big GEMM: M = B*T = 16384 (6144 blocks)
        x_all_prep<<<32768, 256, 0, stream>>>(tgt, sos, x_all);
        gemm_bf16_wmma<<<dim3(G3_ / BN, (B_ * T_) / BM), NT, 0, stream>>>(
            x_all, E_, Wih_bf, b_ih, gx_all, G3_, E_);
    }

    for (int t = 0; t < T_; ++t) {
        const float* gx_ptr;
        int gx_stride;
        if (hoist) {
            gx_ptr    = gx_all + (size_t)t * G3_;
            gx_stride = T_ * G3_;
        } else {
            x_prep<<<(B_ * E_) / 256, 256, 0, stream>>>(tgt, sos, x_bf, t);
            gemm_bf16_wmma<<<dim3(G3_ / BN, B_ / BM), NT, 0, stream>>>(
                x_bf, E_, Wih_bf, b_ih, gx, G3_, E_);
            gx_ptr    = gx;
            gx_stride = G3_;
        }
        // gh = h @ W_hh^T + b_hh   (K = H)
        gemm_bf16_wmma<<<dim3(G3_ / BN, B_ / BM), NT, 0, stream>>>(
            h_bf, H_, Whh_bf, b_hh, gh, G3_, H_);
        // h' = (1-z)*n + z*h
        gru_step<<<(B_ * H_) / 256, 256, 0, stream>>>(gx_ptr, gx_stride, gh, h, h_bf);
        // logits -> d_out[:, t, :]
        gemm_bf16_wmma<<<dim3(E_ / BN, B_ / BM), NT, 0, stream>>>(
            h_bf, H_, Wout_bf, b_out, out + (size_t)t * E_, T_ * E_, H_);
        log_softmax_row<<<B_, 256, 0, stream>>>(out, t);
    }
}
